// GraphNetworkBlock_62534723830204
// MI455X (gfx1250) — matrix operand order repaired and checked by reading, not compiled, not run on hardware
//
#include <hip/hip_runtime.h>

#define N_NODES 50000
#define N_EDGES 1600000
#define LN_EPS  1e-5f

typedef __attribute__((ext_vector_type(16))) _Float16 v16h;
typedef __attribute__((ext_vector_type(8)))  float    v8f;

__device__ __forceinline__ v8f wmma16x16x32(v16h a, v16h b, v8f c) {
  // D = A(16x32 f16) * B(32x16 f16) + C(16x16 f32)
  return __builtin_amdgcn_wmma_f32_16x16x32_f16(
      /*neg_a=*/false, a, /*neg_b=*/false, b,
      /*c_mod=*/(short)0, c, /*reuse_a=*/false, /*reuse_b=*/false);
}

// ---------------------------------------------------------------- zero ws
__global__ void zero_ws_kernel(float* __restrict__ ws, long n) {
  long i = (long)blockIdx.x * blockDim.x + threadIdx.x;
  long stride = (long)gridDim.x * blockDim.x;
  for (; i < n; i += stride) ws[i] = 0.0f;
}

// ---------------------------------------------------------------- edge MLP
// phi_e([e(16), x_src(32), x_dst(32), u(16)]) : 96 -> 128 -> 16, ReLU, LN
// 8 waves/block, 16 edges per wave (128 edges/block).
// Weights staged in LDS pre-swizzled into per-lane B-fragment order:
//   frag(t,c), lane ln, elem j  <-  W[(32c + (ln>>4)*8 + (j>>3)*16 + (j&7))*NOUT + 16t + (ln&15)]
// so each B operand is one contiguous 32 B per-lane load.
__global__ __launch_bounds__(256) void edge_mlp_kernel(
    const float* __restrict__ x, const float* __restrict__ ea_g,
    const float* __restrict__ u, const int* __restrict__ src,
    const int* __restrict__ dst,
    const float* __restrict__ W1, const float* __restrict__ b1,
    const float* __restrict__ W2, const float* __restrict__ b2,
    const float* __restrict__ gam, const float* __restrict__ bet,
    float* __restrict__ e_out, float* __restrict__ agg,
    float* __restrict__ cnt, float* __restrict__ esum)
{
  __shared__ __align__(32) _Float16 sW1f[8 * 3 * 32 * 16];   // 24 KB, frag order
  __shared__ __align__(32) _Float16 sW2f[4 * 32 * 16];       // 4 KB, frag order
  __shared__ float    sB1[128];
  __shared__ float    sB2[16], sG[16], sBt[16];
  __shared__ __align__(16) _Float16 sH[8 * 16 * 128];        // 32 KB wave scratch
  __shared__ float    sESum[16];

  const int tid = threadIdx.x;
  // --- stage W1 (96x128) into fragment order: i = ((t*3+c)*32+ln)*16+j
  for (int i = tid; i < 8 * 3 * 32 * 16; i += 256) {
    const int j  = i & 15;
    const int ln = (i >> 4) & 31;
    const int tc = i >> 9;          // 0..23
    const int c  = tc % 3;
    const int t  = tc / 3;
    sW1f[i] = (_Float16)W1[(32 * c + (ln >> 4) * 8 + (j >> 3) * 16 + (j & 7)) * 128 + 16 * t + (ln & 15)];
  }
  // --- stage W2 (128x16) into fragment order: i = (c*32+ln)*16+j
  for (int i = tid; i < 4 * 32 * 16; i += 256) {
    const int j  = i & 15;
    const int ln = (i >> 4) & 31;
    const int c  = i >> 9;          // 0..3
    sW2f[i] = (_Float16)W2[(32 * c + (ln >> 4) * 8 + (j >> 3) * 16 + (j & 7)) * 16 + (ln & 15)];
  }
  if (tid < 128) sB1[tid] = b1[tid];
  if (tid < 16) { sB2[tid] = b2[tid]; sG[tid] = gam[tid]; sBt[tid] = bet[tid]; sESum[tid] = 0.0f; }
  __syncthreads();

  const int lane = tid & 31;
  const int wave = tid >> 5;
  const int m  = lane & 15;   // row within 16-edge tile
  const int kh = lane >> 4;   // K half-group select

  const long e     = (long)blockIdx.x * 128 + wave * 16 + m;
  const bool valid = (e < N_EDGES);
  const long ec    = valid ? e : (long)(N_EDGES - 1);
  const int  dIdx  = dst[ec];
  const float* eaRow = ea_g + ec * 16;
  const float* sxRow = x + (long)src[ec] * 32;
  const float* dxRow = x + (long)dIdx * 32;

  // each 8-run of the 96-wide concat input lies wholly inside one source
  auto runPtr = [&](int kb) -> const float* {
    if (kb < 16) return eaRow + kb;
    if (kb < 48) return sxRow + (kb - 16);
    if (kb < 80) return dxRow + (kb - 48);
    return u + (kb - 80);
  };

  // ---- A fragments (16x32 f16, ISA layout): halves 0..7 = K kh*8+j,
  //      halves 8..15 = K 16+kh*8+j (within each 32-K chunk)
  v16h A[3];
#pragma unroll
  for (int c = 0; c < 3; ++c) {
    const float* p0 = runPtr(32 * c + kh * 8);
    const float* p1 = runPtr(32 * c + 16 + kh * 8);
#pragma unroll
    for (int j = 0; j < 8; ++j) {
      A[c][j]     = (_Float16)p0[j];
      A[c][8 + j] = (_Float16)p1[j];
    }
  }

  const v16h* BW1 = (const v16h*)sW1f;
  const v16h* BW2 = (const v16h*)sW2f;

  // ---- layer 1: 96 -> 128 (3 K-chunks x 8 col-tiles = 24 WMMA)
  v8f acc[8];
#pragma unroll
  for (int t = 0; t < 8; ++t)
#pragma unroll
    for (int r = 0; r < 8; ++r) acc[t][r] = 0.0f;

#pragma unroll
  for (int c = 0; c < 3; ++c) {       // chunk-outer: A[c] feeds 8 WMMAs in a row
#pragma unroll
    for (int t = 0; t < 8; ++t) {
      v16h B = BW1[(t * 3 + c) * 32 + lane];
      acc[t] = wmma16x16x32(A[c], B, acc[t]);
    }
  }

  // ---- bias + ReLU, C-layout -> LDS (row-major f16) for layer-2 A
  _Float16* H = sH + wave * (16 * 128);
#pragma unroll
  for (int t = 0; t < 8; ++t) {
#pragma unroll
    for (int r = 0; r < 8; ++r) {
      float h = fmaxf(acc[t][r] + sB1[16 * t + m], 0.0f);
      H[(r + 8 * kh) * 128 + 16 * t + m] = (_Float16)h;
    }
  }

  v16h A2[4];
#pragma unroll
  for (int c = 0; c < 4; ++c) {
    const int kb0 = 32 * c + kh * 8;
    const int kb1 = 32 * c + 16 + kh * 8;
#pragma unroll
    for (int j = 0; j < 8; ++j) {
      A2[c][j]     = H[m * 128 + kb0 + j];
      A2[c][8 + j] = H[m * 128 + kb1 + j];
    }
  }

  // ---- layer 2: 128 -> 16 (4 WMMA)
  v8f acc2;
#pragma unroll
  for (int r = 0; r < 8; ++r) acc2[r] = 0.0f;
#pragma unroll
  for (int c = 0; c < 4; ++c) {
    v16h B = BW2[c * 32 + lane];
    acc2 = wmma16x16x32(A2[c], B, acc2);
  }

  // ---- bias + ReLU in fp32, stash 16x16 tile in (reused) wave scratch
  float* OutT = (float*)H;
#pragma unroll
  for (int r = 0; r < 8; ++r)
    OutT[(r + 8 * kh) * 16 + m] = fmaxf(acc2[r] + sB2[m], 0.0f);

  // ---- LayerNorm per row (lanes 0..15 own rows 0..15), store + scatter
  if (lane < 16) {
    float v[16];
    float mu = 0.0f;
#pragma unroll
    for (int j = 0; j < 16; ++j) { v[j] = OutT[lane * 16 + j]; mu += v[j]; }
    mu *= (1.0f / 16.0f);
    float var = 0.0f;
#pragma unroll
    for (int j = 0; j < 16; ++j) { float d = v[j] - mu; var += d * d; }
    var *= (1.0f / 16.0f);
    const float rstd = __frsqrt_rn(var + LN_EPS);
    if (valid) {
      float* orow = e_out + e * 16;
      float* arow = agg + (long)dIdx * 16;
#pragma unroll
      for (int j = 0; j < 16; ++j) {
        const float o = (v[j] - mu) * rstd * sG[j] + sBt[j];
        orow[j] = o;
        atomicAdd(&arow[j], o);       // scatter-sum for NodeModel
        atomicAdd(&sESum[j], o);      // block-local column sum
      }
      atomicAdd(&cnt[dIdx], 1.0f);
    }
  }
  __syncthreads();
  if (tid < 16) atomicAdd(&esum[tid], sESum[tid]);
}

// ---------------------------------------------------------------- node MLP
// phi_n([x(32), agg(16), u(16)]) : 64 -> 128 -> 32, ReLU, LN
__global__ __launch_bounds__(256) void node_mlp_kernel(
    const float* __restrict__ x, const float* __restrict__ u,
    const float* __restrict__ agg, const float* __restrict__ cnt,
    const float* __restrict__ W1, const float* __restrict__ b1,
    const float* __restrict__ W2, const float* __restrict__ b2,
    const float* __restrict__ gam, const float* __restrict__ bet,
    float* __restrict__ x_out, float* __restrict__ xsum)
{
  __shared__ __align__(32) _Float16 sW1f[8 * 2 * 32 * 16];   // 16 KB, frag order
  __shared__ __align__(32) _Float16 sW2f[2 * 4 * 32 * 16];   // 8 KB, frag order
  __shared__ float    sB1[128];
  __shared__ float    sB2[32], sG[32], sBt[32];
  __shared__ __align__(16) _Float16 sH[8 * 16 * 128];        // 32 KB wave scratch
  __shared__ float    sXSum[32];

  const int tid = threadIdx.x;
  // --- stage W1 (64x128): i = ((t*2+c)*32+ln)*16+j
  for (int i = tid; i < 8 * 2 * 32 * 16; i += 256) {
    const int j  = i & 15;
    const int ln = (i >> 4) & 31;
    const int tc = i >> 9;          // 0..15
    const int c  = tc & 1;
    const int t  = tc >> 1;
    sW1f[i] = (_Float16)W1[(32 * c + (ln >> 4) * 8 + (j >> 3) * 16 + (j & 7)) * 128 + 16 * t + (ln & 15)];
  }
  // --- stage W2 (128x32): i = ((t2*4+c)*32+ln)*16+j
  for (int i = tid; i < 2 * 4 * 32 * 16; i += 256) {
    const int j  = i & 15;
    const int ln = (i >> 4) & 31;
    const int tc = i >> 9;          // 0..7
    const int c  = tc & 3;
    const int t2 = tc >> 2;
    sW2f[i] = (_Float16)W2[(32 * c + (ln >> 4) * 8 + (j >> 3) * 16 + (j & 7)) * 32 + 16 * t2 + (ln & 15)];
  }
  if (tid < 128) sB1[tid] = b1[tid];
  if (tid < 32) { sB2[tid] = b2[tid]; sG[tid] = gam[tid]; sBt[tid] = bet[tid]; sXSum[tid] = 0.0f; }
  __syncthreads();

  const int lane = tid & 31;
  const int wave = tid >> 5;
  const int m  = lane & 15;
  const int kh = lane >> 4;

  const long n     = (long)blockIdx.x * 128 + wave * 16 + m;
  const bool valid = (n < N_NODES);
  const long nc    = valid ? n : (long)(N_NODES - 1);
  const float* xr = x + nc * 32;
  const float* ar = agg + nc * 16;
  const float inv = 1.0f / fmaxf(cnt[nc], 1.0f);   // scatter-mean divisor

  auto fetchN = [&](int k) -> float {
    if (k < 32) return xr[k];
    if (k < 48) return ar[k - 32] * inv;
    return u[k - 48];
  };

  v16h A[2];
#pragma unroll
  for (int c = 0; c < 2; ++c) {
    const int kb0 = 32 * c + kh * 8;
    const int kb1 = 32 * c + 16 + kh * 8;
#pragma unroll
    for (int j = 0; j < 8; ++j) {
      A[c][j]     = (_Float16)fetchN(kb0 + j);
      A[c][8 + j] = (_Float16)fetchN(kb1 + j);
    }
  }

  const v16h* BW1 = (const v16h*)sW1f;
  const v16h* BW2 = (const v16h*)sW2f;

  // ---- layer 1: 64 -> 128 (2 chunks x 8 tiles = 16 WMMA)
  v8f acc[8];
#pragma unroll
  for (int t = 0; t < 8; ++t)
#pragma unroll
    for (int r = 0; r < 8; ++r) acc[t][r] = 0.0f;

#pragma unroll
  for (int c = 0; c < 2; ++c) {
#pragma unroll
    for (int t = 0; t < 8; ++t) {
      v16h B = BW1[(t * 2 + c) * 32 + lane];
      acc[t] = wmma16x16x32(A[c], B, acc[t]);
    }
  }

  _Float16* H = sH + wave * (16 * 128);
#pragma unroll
  for (int t = 0; t < 8; ++t) {
#pragma unroll
    for (int r = 0; r < 8; ++r) {
      float h = fmaxf(acc[t][r] + sB1[16 * t + m], 0.0f);
      H[(r + 8 * kh) * 128 + 16 * t + m] = (_Float16)h;
    }
  }

  v16h A2[4];
#pragma unroll
  for (int c = 0; c < 4; ++c) {
    const int kb0 = 32 * c + kh * 8;
    const int kb1 = 32 * c + 16 + kh * 8;
#pragma unroll
    for (int j = 0; j < 8; ++j) {
      A2[c][j]     = H[m * 128 + kb0 + j];
      A2[c][8 + j] = H[m * 128 + kb1 + j];
    }
  }

  // ---- layer 2: 128 -> 32 (4 chunks x 2 tiles = 8 WMMA)
  v8f acc2[2];
#pragma unroll
  for (int t = 0; t < 2; ++t)
#pragma unroll
    for (int r = 0; r < 8; ++r) acc2[t][r] = 0.0f;
#pragma unroll
  for (int c = 0; c < 4; ++c) {
#pragma unroll
    for (int t = 0; t < 2; ++t) {
      v16h B = BW2[(t * 4 + c) * 32 + lane];
      acc2[t] = wmma16x16x32(A2[c], B, acc2[t]);
    }
  }

  float* OutT = (float*)H;   // 16x32 fp32 tile (2 KB) in 4 KB scratch
#pragma unroll
  for (int t = 0; t < 2; ++t)
#pragma unroll
    for (int r = 0; r < 8; ++r)
      OutT[(r + 8 * kh) * 32 + 16 * t + m] = fmaxf(acc2[t][r] + sB2[16 * t + m], 0.0f);

  if (lane < 16) {
    float v[32];
    float mu = 0.0f;
#pragma unroll
    for (int j = 0; j < 32; ++j) { v[j] = OutT[lane * 32 + j]; mu += v[j]; }
    mu *= (1.0f / 32.0f);
    float var = 0.0f;
#pragma unroll
    for (int j = 0; j < 32; ++j) { float d = v[j] - mu; var += d * d; }
    var *= (1.0f / 32.0f);
    const float rstd = __frsqrt_rn(var + LN_EPS);
    if (valid) {
      float* orow = x_out + n * 32;
#pragma unroll
      for (int j = 0; j < 32; ++j) {
        const float o = (v[j] - mu) * rstd * sG[j] + sBt[j];
        orow[j] = o;
        atomicAdd(&sXSum[j], o);
      }
    }
  }
  __syncthreads();
  if (tid < 32) atomicAdd(&xsum[tid], sXSum[tid]);
}

// ---------------------------------------------------------------- global MLP
// phi_g([u(16), mean(x_out)(32), mean(e_out)(16)]) : 64 -> 128 -> 16 (ReLU both)
__global__ __launch_bounds__(128) void global_mlp_kernel(
    const float* __restrict__ u, const float* __restrict__ xsum,
    const float* __restrict__ esum,
    const float* __restrict__ W1, const float* __restrict__ b1,
    const float* __restrict__ W2, const float* __restrict__ b2,
    float* __restrict__ u_out)
{
  __shared__ float gin[64];
  __shared__ float h[128];
  const int tid = threadIdx.x;   // 128 threads
  if (tid < 16)       gin[tid] = u[tid];
  else if (tid < 48)  gin[tid] = xsum[tid - 16] * (1.0f / (float)N_NODES);
  else if (tid < 64)  gin[tid] = esum[tid - 48] * (1.0f / (float)N_EDGES);
  __syncthreads();
  {
    float a = b1[tid];
    for (int k = 0; k < 64; ++k) a += gin[k] * W1[k * 128 + tid];
    h[tid] = fmaxf(a, 0.0f);
  }
  __syncthreads();
  if (tid < 16) {
    float a = b2[tid];
    for (int k = 0; k < 128; ++k) a += h[k] * W2[k * 16 + tid];
    u_out[tid] = fmaxf(a, 0.0f);
  }
}

// ---------------------------------------------------------------- launch
extern "C" void kernel_launch(void* const* d_in, const int* in_sizes, int n_in,
                              void* d_out, int out_size, void* d_ws, size_t ws_size,
                              hipStream_t stream) {
  const float* x         = (const float*)d_in[0];
  const float* edge_attr = (const float*)d_in[1];
  const float* u         = (const float*)d_in[2];
  const int*   src       = (const int*)d_in[3];
  const int*   dst       = (const int*)d_in[4];
  const float* e_W1 = (const float*)d_in[5];
  const float* e_b1 = (const float*)d_in[6];
  const float* e_W2 = (const float*)d_in[7];
  const float* e_b2 = (const float*)d_in[8];
  const float* e_g  = (const float*)d_in[9];
  const float* e_bt = (const float*)d_in[10];
  const float* n_W1 = (const float*)d_in[11];
  const float* n_b1 = (const float*)d_in[12];
  const float* n_W2 = (const float*)d_in[13];
  const float* n_b2 = (const float*)d_in[14];
  const float* n_g  = (const float*)d_in[15];
  const float* n_bt = (const float*)d_in[16];
  const float* g_W1 = (const float*)d_in[17];
  const float* g_b1 = (const float*)d_in[18];
  const float* g_W2 = (const float*)d_in[19];
  const float* g_b2 = (const float*)d_in[20];

  float* out   = (float*)d_out;
  float* x_out = out;
  float* e_out = out + (long)N_NODES * 32;
  float* u_out = e_out + (long)N_EDGES * 16;

  float* ws   = (float*)d_ws;
  float* agg  = ws;                         // N_NODES*16
  float* cnt  = agg + (long)N_NODES * 16;   // N_NODES
  float* esum = cnt + N_NODES;              // 16
  float* xsum = esum + 16;                  // 32
  const long zn = (long)N_NODES * 16 + N_NODES + 16 + 32;

  zero_ws_kernel<<<512, 256, 0, stream>>>(ws, zn);

  edge_mlp_kernel<<<(N_EDGES + 127) / 128, 256, 0, stream>>>(
      x, edge_attr, u, src, dst,
      e_W1, e_b1, e_W2, e_b2, e_g, e_bt,
      e_out, agg, cnt, esum);

  node_mlp_kernel<<<(N_NODES + 127) / 128, 256, 0, stream>>>(
      x, u, agg, cnt,
      n_W1, n_b1, n_W2, n_b2, n_g, n_bt,
      x_out, xsum);

  global_mlp_kernel<<<1, 128, 0, stream>>>(
      u, xsum, esum, g_W1, g_b1, g_W2, g_b2, u_out);
}
